// GATWithLinkPrediction_6777458393292
// MI455X (gfx1250) — matrix operand order repaired and checked by reading, not compiled, not run on hardware
//
#include <hip/hip_runtime.h>
#include <math.h>

typedef __attribute__((ext_vector_type(16))) _Float16 v16h;
typedef __attribute__((ext_vector_type(8)))  _Float16 v8h;
typedef __attribute__((ext_vector_type(8)))  float    v8f;

#define NN      8192
#define HEADS   4
#define TOPK    7
#define NEDGE_K (NN * TOPK)          // 57344
#define NEDGE   (NEDGE_K + NN)       // 65536
#define BN_INV  0.9999950000374994f  // 1/sqrt(1+1e-5)

enum { EP_NONE = 0, EP_BIAS_BN_ELU = 1, EP_BIAS_BN = 2, EP_BIAS_ELU_BN = 3 };

__device__ __forceinline__ float eluf(float v)  { return v > 0.f ? v : __expf(v) - 1.f; }
__device__ __forceinline__ float lrelu(float v) { return v > 0.f ? v : 0.2f * v; }

// ---------------------------------------------------------------------------
// Precision staging: activations and weights converted to f16 once, so the
// WMMA loop is pure f16 b128 loads + v_wmma (no per-tile cvt, no redundancy).
// Weights stored TRANSPOSED (column-major, K-contiguous) so a B fragment is a
// pair of contiguous 16B reads (k0+8h, k0+8h+16) matching the ISA 7.12.2 B layout.
// ---------------------------------------------------------------------------
__global__ void cvt_f16(const float* __restrict__ in, _Float16* __restrict__ out,
                        int n) {
  int t = blockIdx.x * blockDim.x + threadIdx.x;
  if (t < n) out[t] = (_Float16)in[t];
}

template <int K, int N>
__global__ void cvt_wt(const float* __restrict__ W, _Float16* __restrict__ Wt) {
  int t = blockIdx.x * blockDim.x + threadIdx.x;
  if (t >= K * N) return;
  int k = t / N, n = t - k * N;           // coalesced read of W[k][n]
  Wt[(size_t)n * K + k] = (_Float16)W[t]; // Wt[n][k]
}

// ---------------------------------------------------------------------------
// Wave-level WMMA GEMM: out[8192, NDIM] = epi(A[8192,KDIM] @ W[KDIM,NDIM])
// One wave computes a 16 x (16*NT) strip: A fragment loaded once per k-step,
// reused across NT independent v_wmma_f32_16x16x32_f16 accumulators.
// Layouts per CDNA5 ISA 7.12.2:
//   A 16x32 f16 : lane<16 row=lane, halves = K{0..7,16..23}; lane>=16 K{8..15,24..31}
//   B 32x16 f16 : col = lane&15, same K halves as A: lane<16 K{0..7,16..23}; lane>=16 K{8..15,24..31}
//   C 16x16 f32 : vgpr v -> row = v + 8*(lane>>4), col = lane&15
// ---------------------------------------------------------------------------
template <int KDIM, int NDIM, int NT, int EP>
__global__ __launch_bounds__(32) void wmma_gemm(const _Float16* __restrict__ A,
                                                const _Float16* __restrict__ Wt,
                                                const float* __restrict__ bias,
                                                const float* __restrict__ gamma,
                                                const float* __restrict__ beta,
                                                float* __restrict__ out) {
  const int lane    = threadIdx.x;
  const int halfsel = lane >> 4;
  const int l16     = lane & 15;
  const int tm      = blockIdx.y;
  const int n0      = blockIdx.x * (16 * NT);  // first column of this wave's strip
  const int row     = tm * 16 + l16;

  v8f acc[NT] = {};
#pragma unroll
  for (int k0 = 0; k0 < KDIM; k0 += 32) {
    // A fragment: two contiguous 16B chunks per lane
    const _Float16* ar = A + (size_t)row * KDIM + k0 + 8 * halfsel;
    v8h alo = *(const v8h*)(ar);
    v8h ahi = *(const v8h*)(ar + 16);
    v16h a  = __builtin_shufflevector(alo, ahi, 0, 1, 2, 3, 4, 5, 6, 7,
                                      8, 9, 10, 11, 12, 13, 14, 15);
#pragma unroll
    for (int t = 0; t < NT; ++t) {
      const int col = n0 + t * 16 + l16;
      // B fragment: two contiguous 16B chunks per lane, same K order as A
      const _Float16* wr = Wt + (size_t)col * KDIM + k0 + 8 * halfsel;
      v8h blo = *(const v8h*)(wr);
      v8h bhi = *(const v8h*)(wr + 16);
      v16h b  = __builtin_shufflevector(blo, bhi, 0, 1, 2, 3, 4, 5, 6, 7,
                                        8, 9, 10, 11, 12, 13, 14, 15);
      acc[t] = __builtin_amdgcn_wmma_f32_16x16x32_f16(false, a, false, b,
                                                      (short)0, acc[t], false,
                                                      false);
    }
  }

#pragma unroll
  for (int t = 0; t < NT; ++t) {
    const int col = n0 + t * 16 + l16;
#pragma unroll
    for (int v = 0; v < 8; ++v) {
      const int r = tm * 16 + v + 8 * halfsel;
      float val   = acc[t][v];
      if (EP == EP_BIAS_BN_ELU) {
        val = eluf(gamma[col] * (val + bias[col]) * BN_INV + beta[col]);
      } else if (EP == EP_BIAS_BN) {
        val = gamma[col] * (val + bias[col]) * BN_INV + beta[col];
      } else if (EP == EP_BIAS_ELU_BN) {
        val = gamma[col] * eluf(val + bias[col]) * BN_INV + beta[col];
      }
      out[(size_t)r * NDIM + col] = val;
    }
  }
}

// ---------------------------------------------------------------------------
// embA[n,d] = sum_k emb[n,k] * lp_A[k,d]   (tiny 16x16 weight)
// ---------------------------------------------------------------------------
__global__ void bilinear_pre(const float* __restrict__ emb,
                             const float* __restrict__ lpA,
                             float* __restrict__ embA) {
  int t = blockIdx.x * blockDim.x + threadIdx.x;
  if (t >= NN * 16) return;
  int n = t >> 4, d = t & 15;
  float acc = 0.f;
#pragma unroll
  for (int k = 0; k < 16; ++k) acc += emb[n * 16 + k] * lpA[k * 16 + d];
  embA[t] = acc;
}

// ---------------------------------------------------------------------------
// One wave per row i: recompute scores against all j (emb/embA stay L2-hot),
// write upper-triangle link_logits, keep per-lane top-7, merge in LDS.
// M[i,j] = (j>i) ? embA[i].emb[j]+b : embA[j].emb[i]+b, diag excluded.
// ---------------------------------------------------------------------------
__global__ __launch_bounds__(32) void link_topk_kernel(
    const float* __restrict__ emb, const float* __restrict__ embA,
    const float* __restrict__ lp_b, float* __restrict__ link_logits,
    int* __restrict__ sel) {
  const int i = blockIdx.x;
  const int lane = threadIdx.x;
  __shared__ float ea[16], ei[16];
  __shared__ float sv[32 * TOPK];
  __shared__ int   si[32 * TOPK];
  if (lane < 16) { ea[lane] = embA[i * 16 + lane]; ei[lane] = emb[i * 16 + lane]; }
  __syncthreads();
  const float b = lp_b[0];
  const long long tri = (long long)i * NN - (long long)i * (i + 1) / 2 - i - 1;

  float tv[TOPK]; int ti[TOPK];
#pragma unroll
  for (int k = 0; k < TOPK; ++k) { tv[k] = -3e38f; ti[k] = -1; }

  for (int j = lane; j < NN; j += 32) {
    if (j == i) continue;
    float s = b;
    if (j > i) {
      const float* ej = emb + (size_t)j * 16;
#pragma unroll
      for (int d = 0; d < 16; ++d) s += ea[d] * ej[d];
      link_logits[tri + j] = s;
    } else {
      const float* aj = embA + (size_t)j * 16;
#pragma unroll
      for (int d = 0; d < 16; ++d) s += aj[d] * ei[d];
    }
    if (s > tv[TOPK - 1]) {
      int p = TOPK - 1;
      while (p > 0 && tv[p - 1] < s) { tv[p] = tv[p - 1]; ti[p] = ti[p - 1]; --p; }
      tv[p] = s; ti[p] = j;
    }
  }
#pragma unroll
  for (int k = 0; k < TOPK; ++k) { sv[lane * TOPK + k] = tv[k]; si[lane * TOPK + k] = ti[k]; }
  __syncthreads();
  if (lane == 0) {
    for (int k = 0; k < TOPK; ++k) {
      float best = -3e38f; int bi = 0, bp = 0;
      for (int t = 0; t < 32 * TOPK; ++t)
        if (sv[t] > best) { best = sv[t]; bi = si[t]; bp = t; }
      sv[bp] = -3e38f;
      sel[i * TOPK + k] = bi;
    }
  }
}

// ---------------------------------------------------------------------------
// Canonical edges + dedup mask + self loops; also writes edge_index output.
// ---------------------------------------------------------------------------
__global__ void build_edges(const int* __restrict__ sel, int* __restrict__ srcA,
                            int* __restrict__ dstA, int* __restrict__ validA,
                            int* __restrict__ edge_out) {
  int e = blockIdx.x * blockDim.x + threadIdx.x;
  if (e >= NEDGE) return;
  int s, d, v;
  if (e < NEDGE_K) {
    int i = e / TOPK;
    int j = sel[e];
    int dup = 0;
    if (j < i) {
#pragma unroll
      for (int kk = 0; kk < TOPK; ++kk) dup |= (sel[j * TOPK + kk] == i);
    }
    s = j < i ? j : i;
    d = j < i ? i : j;
    v = !dup;
  } else {
    int i = e - NEDGE_K;
    s = i; d = i; v = 1;
  }
  srcA[e] = s; dstA[e] = d; validA[e] = v;
  edge_out[e] = s; edge_out[NEDGE + e] = d;
}

// ---------------------------------------------------------------------------
// GAT sparse phase (E = 65536 edges -> scalar + atomics is right-sized)
// ---------------------------------------------------------------------------
__device__ __forceinline__ void atomicMaxF(float* addr, float val) {
  int* ai = (int*)addr;
  int old = *ai;
  while (__int_as_float(old) < val) {
    int assumed = old;
    old = atomicCAS(ai, assumed, __float_as_int(val));
    if (old == assumed) break;
  }
}

__global__ void node_att(const float* __restrict__ h, const float* __restrict__ a_src,
                         const float* __restrict__ a_dst, float* __restrict__ asrc,
                         float* __restrict__ adst) {
  int t = blockIdx.x * blockDim.x + threadIdx.x;  // node*HEADS + head
  if (t >= NN * HEADS) return;
  int n = t >> 2, hd = t & 3;
  const float* hp = h + (size_t)n * 512 + hd * 128;
  const float* as = a_src + hd * 128;
  const float* ad = a_dst + hd * 128;
  float sa = 0.f, sd = 0.f;
  for (int d = 0; d < 128; ++d) { float x = hp[d]; sa += x * as[d]; sd += x * ad[d]; }
  asrc[t] = sa; adst[t] = sd;
}

__global__ void init_stats(float* __restrict__ mbuf, float* __restrict__ den,
                           float* __restrict__ agg) {
  int t = blockIdx.x * blockDim.x + threadIdx.x;
  if (t < NN * HEADS) { mbuf[t] = -1e30f; den[t] = 0.f; }
  if (t < NN * 128) agg[t] = 0.f;
}

__global__ void att_logits(const int* __restrict__ srcA, const int* __restrict__ dstA,
                           const int* __restrict__ validA, const float* __restrict__ asrc,
                           const float* __restrict__ adst, float* __restrict__ albuf,
                           float* __restrict__ mbuf) {
  int t = blockIdx.x * blockDim.x + threadIdx.x;
  if (t >= NEDGE * HEADS) return;
  int e = t >> 2, hd = t & 3;
  if (!validA[e]) { albuf[t] = 0.f; return; }
  float v = lrelu(asrc[srcA[e] * HEADS + hd] + adst[dstA[e] * HEADS + hd]);
  albuf[t] = v;
  atomicMaxF(&mbuf[dstA[e] * HEADS + hd], v);
}

__global__ void att_exp(const int* __restrict__ dstA, const int* __restrict__ validA,
                        const float* __restrict__ mbuf, float* __restrict__ albuf,
                        float* __restrict__ den) {
  int t = blockIdx.x * blockDim.x + threadIdx.x;
  if (t >= NEDGE * HEADS) return;
  int e = t >> 2, hd = t & 3;
  if (!validA[e]) return;
  float ex = __expf(albuf[t] - mbuf[dstA[e] * HEADS + hd]);
  albuf[t] = ex;
  atomicAdd(&den[dstA[e] * HEADS + hd], ex);
}

__global__ void att_agg(const int* __restrict__ srcA, const int* __restrict__ dstA,
                        const int* __restrict__ validA, const float* __restrict__ albuf,
                        const float* __restrict__ den, const float* __restrict__ h,
                        float* __restrict__ agg) {
  int t = blockIdx.x * blockDim.x + threadIdx.x;
  if (t >= NEDGE * HEADS) return;
  int e = t >> 2, hd = t & 3;
  if (!validA[e]) return;
  float c = albuf[t] / den[dstA[e] * HEADS + hd] * (1.f / HEADS);
  const float* hp = h + (size_t)srcA[e] * 512 + hd * 128;
  float* ap = agg + (size_t)dstA[e] * 128;
  for (int d = 0; d < 128; ++d) atomicAdd(&ap[d], c * hp[d]);
}

__global__ void bias_bn_elu(const float* __restrict__ agg, const float* __restrict__ b,
                            const float* __restrict__ g, const float* __restrict__ be,
                            float* __restrict__ outp) {
  int t = blockIdx.x * blockDim.x + threadIdx.x;
  if (t >= NN * 128) return;
  int d = t & 127;
  outp[t] = eluf(g[d] * (agg[t] + b[d]) * BN_INV + be[d]);
}

// ---------------------------------------------------------------------------
extern "C" void kernel_launch(void* const* d_in, const int* in_sizes, int n_in,
                              void* d_out, int out_size, void* d_ws, size_t ws_size,
                              hipStream_t stream) {
  (void)in_sizes; (void)n_in; (void)out_size; (void)ws_size;
  const float* x     = (const float*)d_in[0];
  const float* il_w1 = (const float*)d_in[1];
  const float* il_b1 = (const float*)d_in[2];
  const float* il_g1 = (const float*)d_in[3];
  const float* il_be1= (const float*)d_in[4];
  const float* il_w2 = (const float*)d_in[5];
  const float* il_b2 = (const float*)d_in[6];
  const float* il_g2 = (const float*)d_in[7];
  const float* il_be2= (const float*)d_in[8];
  const float* lp_A  = (const float*)d_in[9];
  const float* lp_b  = (const float*)d_in[10];
  const float* g0_w  = (const float*)d_in[11];
  const float* g0_as = (const float*)d_in[12];
  const float* g0_ad = (const float*)d_in[13];
  const float* g0_b  = (const float*)d_in[14];
  const float* bn0_g = (const float*)d_in[15];
  const float* bn0_b = (const float*)d_in[16];
  const float* g1_w  = (const float*)d_in[17];
  const float* g1_as = (const float*)d_in[18];
  const float* g1_ad = (const float*)d_in[19];
  const float* g1_b  = (const float*)d_in[20];
  const float* bn1_g = (const float*)d_in[21];
  const float* bn1_b = (const float*)d_in[22];
  const float* f0_w  = (const float*)d_in[23];
  const float* f0_b  = (const float*)d_in[24];
  const float* fbn0_g= (const float*)d_in[25];
  const float* fbn0_b= (const float*)d_in[26];
  const float* f1_w  = (const float*)d_in[27];
  const float* f1_b  = (const float*)d_in[28];
  const float* fbn1_g= (const float*)d_in[29];
  const float* fbn1_b= (const float*)d_in[30];

  // d_out layout: out0 [8192*64] | out1 [8192*16] | edge_index [2*65536] int | link_logits
  float* out0 = (float*)d_out;
  float* out1 = out0 + NN * 64;
  int*   ei   = (int*)(out1 + NN * 16);
  float* ll   = (float*)(ei + 2 * NEDGE);

  // workspace carve-up (256B aligned)
  char* wp = (char*)d_ws;
  auto carve = [&](size_t bytes) -> char* {
    char* p = wp; wp += (bytes + 255) & ~(size_t)255; return p;
  };
  float*    emb    = (float*)   carve(NN * 16 * 4);
  float*    emb1   = (float*)   carve((size_t)NN * 128 * 4);
  float*    embA   = (float*)   carve(NN * 16 * 4);
  int*      sel    = (int*)     carve(NEDGE_K * 4);
  int*      srcA   = (int*)     carve(NEDGE * 4);
  int*      dstA   = (int*)     carve(NEDGE * 4);
  int*      valA   = (int*)     carve(NEDGE * 4);
  float*    hbig   = (float*)   carve((size_t)NN * 512 * 4);
  float*    asrc   = (float*)   carve(NN * HEADS * 4);
  float*    adst   = (float*)   carve(NN * HEADS * 4);
  float*    mbuf   = (float*)   carve(NN * HEADS * 4);
  float*    den    = (float*)   carve(NN * HEADS * 4);
  float*    albuf  = (float*)   carve((size_t)NEDGE * HEADS * 4);
  float*    agg    = (float*)   carve((size_t)NN * 128 * 4);
  float*    gin    = (float*)   carve((size_t)NN * 128 * 4);
  _Float16* xh     = (_Float16*)carve((size_t)NN * 64 * 2);   // f16 x
  _Float16* actH   = (_Float16*)carve((size_t)NN * 128 * 2);  // f16 staging (chained)
  _Float16* il_w1t = (_Float16*)carve(64 * 128 * 2);
  _Float16* il_w2t = (_Float16*)carve(128 * 16 * 2);
  _Float16* g0_wt  = (_Float16*)carve(64 * 512 * 2);
  _Float16* g1_wt  = (_Float16*)carve(128 * 512 * 2);
  _Float16* f0_wt  = (_Float16*)carve(128 * 64 * 2);
  _Float16* f1_wt  = (_Float16*)carve(64 * 16 * 2);

  const dim3 w32(32);
  const int MT = NN / 16;  // 512 row tiles
  const int EH = NEDGE * HEADS;

  // ---- one-time f16 staging of inputs + transposed weights
  cvt_f16<<<(NN * 64 + 255) / 256, 256, 0, stream>>>(x, xh, NN * 64);
  cvt_wt<64, 128><<<(64 * 128 + 255) / 256, 256, 0, stream>>>(il_w1, il_w1t);
  cvt_wt<128, 16><<<(128 * 16 + 255) / 256, 256, 0, stream>>>(il_w2, il_w2t);
  cvt_wt<64, 512><<<(64 * 512 + 255) / 256, 256, 0, stream>>>(g0_w, g0_wt);
  cvt_wt<128, 512><<<(128 * 512 + 255) / 256, 256, 0, stream>>>(g1_w, g1_wt);
  cvt_wt<128, 64><<<(128 * 64 + 255) / 256, 256, 0, stream>>>(f0_w, f0_wt);
  cvt_wt<64, 16><<<(64 * 16 + 255) / 256, 256, 0, stream>>>(f1_w, f1_wt);

  // ---- initial MLP: x -> emb1 -> emb
  wmma_gemm<64, 128, 4, EP_BIAS_BN_ELU><<<dim3(128 / 64, MT), w32, 0, stream>>>(
      xh, il_w1t, il_b1, il_g1, il_be1, emb1);
  cvt_f16<<<(NN * 128 + 255) / 256, 256, 0, stream>>>(emb1, actH, NN * 128);
  wmma_gemm<128, 16, 1, EP_BIAS_BN_ELU><<<dim3(1, MT), w32, 0, stream>>>(
      actH, il_w2t, il_b2, il_g2, il_be2, emb);

  // ---- bilinear scores, link_logits, top-7, edges
  bilinear_pre<<<(NN * 16 + 255) / 256, 256, 0, stream>>>(emb, lp_A, embA);
  link_topk_kernel<<<NN, w32, 0, stream>>>(emb, embA, lp_b, ll, sel);
  build_edges<<<(NEDGE + 255) / 256, 256, 0, stream>>>(sel, srcA, dstA, valA, ei);

  // ---- GAT layer 0
  wmma_gemm<64, 512, 4, EP_NONE><<<dim3(512 / 64, MT), w32, 0, stream>>>(
      xh, g0_wt, nullptr, nullptr, nullptr, hbig);
  node_att<<<(NN * HEADS + 255) / 256, 256, 0, stream>>>(hbig, g0_as, g0_ad, asrc, adst);
  init_stats<<<(NN * 128 + 255) / 256, 256, 0, stream>>>(mbuf, den, agg);
  att_logits<<<(EH + 255) / 256, 256, 0, stream>>>(srcA, dstA, valA, asrc, adst, albuf, mbuf);
  att_exp<<<(EH + 255) / 256, 256, 0, stream>>>(dstA, valA, mbuf, albuf, den);
  att_agg<<<(EH + 255) / 256, 256, 0, stream>>>(srcA, dstA, valA, albuf, den, hbig, agg);
  bias_bn_elu<<<(NN * 128 + 255) / 256, 256, 0, stream>>>(agg, g0_b, bn0_g, bn0_b, gin);

  // ---- GAT layer 1
  cvt_f16<<<(NN * 128 + 255) / 256, 256, 0, stream>>>(gin, actH, NN * 128);
  wmma_gemm<128, 512, 4, EP_NONE><<<dim3(512 / 64, MT), w32, 0, stream>>>(
      actH, g1_wt, nullptr, nullptr, nullptr, hbig);
  node_att<<<(NN * HEADS + 255) / 256, 256, 0, stream>>>(hbig, g1_as, g1_ad, asrc, adst);
  init_stats<<<(NN * 128 + 255) / 256, 256, 0, stream>>>(mbuf, den, agg);
  att_logits<<<(EH + 255) / 256, 256, 0, stream>>>(srcA, dstA, valA, asrc, adst, albuf, mbuf);
  att_exp<<<(EH + 255) / 256, 256, 0, stream>>>(dstA, valA, mbuf, albuf, den);
  att_agg<<<(EH + 255) / 256, 256, 0, stream>>>(srcA, dstA, valA, albuf, den, hbig, agg);
  bias_bn_elu<<<(NN * 128 + 255) / 256, 256, 0, stream>>>(agg, g1_b, bn1_g, bn1_b, gin);

  // ---- head: out0 = BN(h@f0_w + b), out1 = BN(ELU(out0@f1_w + b))
  cvt_f16<<<(NN * 128 + 255) / 256, 256, 0, stream>>>(gin, actH, NN * 128);
  wmma_gemm<128, 64, 4, EP_BIAS_BN><<<dim3(1, MT), w32, 0, stream>>>(
      actH, f0_wt, f0_b, fbn0_g, fbn0_b, out0);
  cvt_f16<<<(NN * 64 + 255) / 256, 256, 0, stream>>>(out0, actH, NN * 64);
  wmma_gemm<64, 16, 1, EP_BIAS_ELU_BN><<<dim3(1, MT), w32, 0, stream>>>(
      actH, f1_wt, f1_b, fbn1_g, fbn1_b, out1);
}
